// MyRNN_82454782148949
// MI455X (gfx1250) — matrix operand order repaired and checked by reading, not compiled, not run on hardware
//
#include <hip/hip_runtime.h>
#include <hip/hip_bf16.h>
#include <math.h>

typedef __bf16 bf16_t;
typedef __attribute__((ext_vector_type(16))) __bf16 v16bf;
typedef __attribute__((ext_vector_type(8)))  __bf16 v8bf;
typedef __attribute__((ext_vector_type(8)))  float  v8f;

#define LDSS 40  // LDS row stride in bf16 elems (80B: 16B-aligned, staggers banks)

#if defined(__HIP_DEVICE_COMPILE__) && __has_builtin(__builtin_amdgcn_global_load_async_to_lds_b128)
#define HAS_ASYNC_LDS 1
#else
#define HAS_ASYNC_LDS 0
#endif

#if HAS_ASYNC_LDS
typedef int v4i_ld __attribute__((__vector_size__(4 * sizeof(int))));
typedef __attribute__((address_space(1))) v4i_ld* g_v4p;
typedef __attribute__((address_space(3))) v4i_ld* l_v4p;
#endif

__device__ __forceinline__ void wait_async_zero() {
#if HAS_ASYNC_LDS
#if __has_builtin(__builtin_amdgcn_s_wait_asynccnt)
  __builtin_amdgcn_s_wait_asynccnt(0);
#else
  asm volatile("s_wait_asynccnt 0" ::: "memory");
#endif
#endif
}

// copy 32 bytes (16 bf16) global -> LDS
__device__ __forceinline__ void ld32B(bf16_t* lds, const bf16_t* gp) {
#if HAS_ASYNC_LDS
  g_v4p gsrc = (g_v4p)gp;   // C-style cast: const drop + addrspacecast to AS1
  l_v4p ldst = (l_v4p)lds;  // addrspacecast generic -> AS3
  __builtin_amdgcn_global_load_async_to_lds_b128(gsrc, ldst, 0, 0);
  __builtin_amdgcn_global_load_async_to_lds_b128(gsrc, ldst, 16, 0);
#else
  *(uint4*)lds       = *(const uint4*)gp;
  *(uint4*)(lds + 8) = *(const uint4*)(gp + 8);
#endif
}

// ---------- small utility kernels ----------
__global__ void k_conv_bf16(const float* __restrict__ src, bf16_t* __restrict__ dst, long n) {
  long i = (long)blockIdx.x * blockDim.x + threadIdx.x;
  if (i < n) dst[i] = (bf16_t)src[i];
}

// dst is [Np][Kp] bf16 = transpose of src [srcK][srcN] f32, zero-padded
__global__ void k_convT_pad(const float* __restrict__ src, bf16_t* __restrict__ dst,
                            int srcK, int srcN, int Kp, int Np) {
  long i = (long)blockIdx.x * blockDim.x + threadIdx.x;
  long total = (long)Kp * Np;
  if (i >= total) return;
  int n = (int)(i / Kp);
  int k = (int)(i % Kp);
  float v = (k < srcK && n < srcN) ? src[(long)k * srcN + n] : 0.0f;
  dst[i] = (bf16_t)v;
}

__global__ void k_pad_f32(const float* __restrict__ src, float* __restrict__ dst, int n, int np) {
  int i = blockIdx.x * blockDim.x + threadIdx.x;
  if (i < np) dst[i] = (i < n) ? src[i] : 0.0f;
}

__global__ void k_zero_bf16(bf16_t* __restrict__ dst, long n) {
  long i = (long)blockIdx.x * blockDim.x + threadIdx.x;
  if (i < n) dst[i] = (bf16_t)0.0f;
}

// ---------- GEMM parameter block ----------
struct GP {
  const bf16_t* A;   long lda;
  const bf16_t* Bt;  long ldb;  int K;
  const bf16_t* A2;  long lda2;
  const bf16_t* B2t; long ldb2; int K2;
  const float* bias;
  const float* pre;  long ldpre;
  float* outF;  long ldoutF; int nvalid;
  bf16_t* outB;  long ldoutB;
  bf16_t* outB2; long ldoutB2;
  int N;     // logical (padded-to-32) output width; store/bias/pre guard
  int ntx;   // #N tiles for this problem (grid.x guard for paired launches)
};

// ---------- generic WMMA GEMM body ----------
// out = act( A[M,K] @ B^T-layout[N,K] (+ A2@B2) + bias + pre )
// B operands pre-transposed to [Ntiles*BN][K] (zero-padded rows), so staging is unguarded.
// M always a multiple of BM by construction. Double-buffered LDS + async global->LDS.
template <int BM, int BN, int WMW, int WNW, int NT, bool TANH>
__device__ __forceinline__ void gemm_body(const GP& g) {
  constexpr int MI = BM / (16 * WMW);
  constexpr int NJ = BN / (16 * WNW);
  __shared__ __align__(16) bf16_t As[2 * BM * LDSS];
  __shared__ __align__(16) bf16_t Bs[2 * BN * LDSS];

  const int tid  = threadIdx.x;
  const int wave = tid >> 5;
  const int lane = tid & 31;
  const int wm   = (wave % WMW) * (MI * 16);
  const int wn   = (wave / WMW) * (NJ * 16);
  const int m16  = lane & 15;
  const int hk   = lane >> 4;
  const long tileM = (long)blockIdx.y * BM;
  const int  tileN = blockIdx.x * BN;

  v8f acc[MI][NJ];
#pragma unroll
  for (int i = 0; i < MI; ++i)
#pragma unroll
    for (int j = 0; j < NJ; ++j)
#pragma unroll
      for (int r = 0; r < 8; ++r) acc[i][j][r] = 0.0f;

  auto stage = [&](int buf, const bf16_t* Ap, long ldA,
                   const bf16_t* Bp, long ldB, int k0) {
    bf16_t* as = As + buf * (BM * LDSS);
    bf16_t* bs = Bs + buf * (BN * LDSS);
    for (int s = tid; s < BM * 2; s += NT) {
      int r = s >> 1, c = (s & 1) * 16;
      ld32B(as + r * LDSS + c, Ap + (tileM + r) * ldA + (k0 + c));
    }
    for (int s = tid; s < BN * 2; s += NT) {
      int r = s >> 1, c = (s & 1) * 16;
      ld32B(bs + r * LDSS + c, Bp + (long)(tileN + r) * ldB + (k0 + c));
    }
  };

  auto compute = [&](int buf) {
    const bf16_t* as = As + buf * (BM * LDSS);
    const bf16_t* bs = Bs + buf * (BN * LDSS);
    v16bf af[MI], bq[NJ];
#pragma unroll
    for (int i = 0; i < MI; ++i) {
      // A frag: row = m16; elems 0..7 -> K=hk*8.., elems 8..15 -> K=16+hk*8..
      const bf16_t* p = &as[(wm + i * 16 + m16) * LDSS + hk * 8];
      v8bf lo = *(const v8bf*)p;
      v8bf hi = *(const v8bf*)(p + 16);
      af[i] = __builtin_shufflevector(lo, hi, 0,1,2,3,4,5,6,7,8,9,10,11,12,13,14,15);
    }
#pragma unroll
    for (int j = 0; j < NJ; ++j) {
      // B frag: col = m16; elems 0..7 -> K=hk*8.., elems 8..15 -> K=16+hk*8..
      const bf16_t* p = &bs[(wn + j * 16 + m16) * LDSS + hk * 8];
      v8bf lo = *(const v8bf*)p;
      v8bf hi = *(const v8bf*)(p + 16);
      bq[j] = __builtin_shufflevector(lo, hi, 0,1,2,3,4,5,6,7,8,9,10,11,12,13,14,15);
    }
#pragma unroll
    for (int i = 0; i < MI; ++i)
#pragma unroll
      for (int j = 0; j < NJ; ++j)
        acc[i][j] = __builtin_amdgcn_wmma_f32_16x16x32_bf16(
            false, af[i], false, bq[j], (short)0, acc[i][j], false, false);
  };

  auto kloop = [&](const bf16_t* Ap, long ldA, const bf16_t* Bp, long ldB, int Kd) {
    const int nk = Kd >> 5;
    __syncthreads();                 // previous LDS consumers done
    stage(0, Ap, ldA, Bp, ldB, 0);   // prologue prefetch
    for (int i = 0; i < nk; ++i) {
      const int cur = i & 1;
      wait_async_zero();             // this wave's async writes for `cur` landed
      __syncthreads();               // everyone's landed; prev compute done
      if (i + 1 < nk) stage(1 - cur, Ap, ldA, Bp, ldB, (i + 1) * 32);
      compute(cur);
    }
  };

  kloop(g.A, g.lda, g.Bt, g.ldb, g.K);
  if (g.K2 > 0) kloop(g.A2, g.lda2, g.B2t, g.ldb2, g.K2);

  // epilogue: C/D layout lane: col = m16, rows = hk*8 + r
#pragma unroll
  for (int i = 0; i < MI; ++i) {
#pragma unroll
    for (int j = 0; j < NJ; ++j) {
      int ng = tileN + wn + j * 16 + m16;
      if (ng >= g.N) continue;
      float bv = g.bias ? g.bias[ng] : 0.0f;
#pragma unroll
      for (int r = 0; r < 8; ++r) {
        long mg = tileM + wm + i * 16 + hk * 8 + r;
        float v = acc[i][j][r] + bv;
        if (g.pre)  v += g.pre[mg * g.ldpre + ng];
        if (TANH)   v = tanhf(v);
        if (g.outF && ng < g.nvalid) g.outF[mg * g.ldoutF + ng] = v;
        if (g.outB)  g.outB[mg * g.ldoutB + ng]   = (bf16_t)v;
        if (g.outB2) g.outB2[mg * g.ldoutB2 + ng] = (bf16_t)v;
      }
    }
  }
}

template <int BM, int BN, int WMW, int WNW, int NT, bool TANH>
__global__ __launch_bounds__(NT) void k_gemm(GP g) {
  if ((int)blockIdx.x >= g.ntx) return;
  gemm_body<BM, BN, WMW, WNW, NT, TANH>(g);
}

// paired launch: z==0 -> g0 (layer-2 step t), z==1 -> g1 (layer-1 step t+1); independent.
template <int BM, int BN, int WMW, int WNW, int NT>
__global__ __launch_bounds__(NT) void k_gemm_pair(GP g0, GP g1) {
  const GP g = (blockIdx.z == 0) ? g0 : g1;
  if ((int)blockIdx.x >= g.ntx) return;
  gemm_body<BM, BN, WMW, WNW, NT, true>(g);
}

// ---------- host ----------
extern "C" void kernel_launch(void* const* d_in, const int* in_sizes, int n_in,
                              void* d_out, int out_size, void* d_ws, size_t ws_size,
                              hipStream_t stream) {
  (void)in_sizes; (void)n_in; (void)out_size; (void)ws_size;
  const float* x  = (const float*)d_in[0];
  const float* W1 = (const float*)d_in[1];
  const float* U1 = (const float*)d_in[2];
  const float* b1 = (const float*)d_in[3];
  const float* W2 = (const float*)d_in[4];
  const float* U2 = (const float*)d_in[5];
  const float* b2 = (const float*)d_in[6];
  const float* Wd = (const float*)d_in[7];
  const float* bd = (const float*)d_in[8];
  float* out = (float*)d_out;

  constexpr int B = 256, T = 200, F = 512, H1 = 521, H2 = 804;
  constexpr int H1p = 544, H2p = 832;     // logical padded widths (mult of 32)
  constexpr int NB_W1 = 640;              // W1^T rows: 5 tiles * 128
  constexpr int NB_U1 = 576;              // U1^T rows: 9 tiles * 64
  constexpr int NB_H2 = 832;              // 13 tiles * 64 (exact)
  const long MT = (long)B * T;            // 51200 (mult of 128)

  // carve workspace
  char* p = (char*)d_ws;
  auto carve = [&](size_t bytes) -> void* {
    char* r = p; p += (bytes + 255) & ~(size_t)255; return (void*)r;
  };
  bf16_t* x_bf  = (bf16_t*)carve((size_t)MT * F * 2);
  bf16_t* W1T   = (bf16_t*)carve((size_t)NB_W1 * F   * 2);
  bf16_t* U1T   = (bf16_t*)carve((size_t)NB_U1 * H1p * 2);
  bf16_t* W2T   = (bf16_t*)carve((size_t)NB_H2 * H1p * 2);
  bf16_t* U2T   = (bf16_t*)carve((size_t)NB_H2 * H2p * 2);
  bf16_t* WdT   = (bf16_t*)carve((size_t)NB_H2 * H2p * 2);
  float*  b1p   = (float*)carve((size_t)H1p * 4);
  float*  b2p   = (float*)carve((size_t)H2p * 4);
  float*  bdp   = (float*)carve((size_t)H2p * 4);
  float*  xW1   = (float*)carve((size_t)MT * H1p * 4);
  bf16_t* h1seq = (bf16_t*)carve((size_t)MT * H1p * 2);
  bf16_t* zero1 = (bf16_t*)carve((size_t)B * H1p * 2);
  bf16_t* h2b0  = (bf16_t*)carve((size_t)B * H2p * 2);
  bf16_t* h2b1  = (bf16_t*)carve((size_t)B * H2p * 2);
  bf16_t* h2buf[2] = { h2b0, h2b1 };

  // --- conversions / padding ---
  {
    long nx = MT * F;
    k_conv_bf16<<<(int)((nx + 255) / 256), 256, 0, stream>>>(x, x_bf, nx);
    auto launchT = [&](const float* s, bf16_t* d, int sk, int sn, int kp, int np_) {
      long tot = (long)kp * np_;
      k_convT_pad<<<(int)((tot + 255) / 256), 256, 0, stream>>>(s, d, sk, sn, kp, np_);
    };
    launchT(W1, W1T, F,  H1, F,   NB_W1);
    launchT(U1, U1T, H1, H1, H1p, NB_U1);
    launchT(W2, W2T, H1, H2, H1p, NB_H2);
    launchT(U2, U2T, H2, H2, H2p, NB_H2);
    launchT(Wd, WdT, H2, H2, H2p, NB_H2);
    k_pad_f32<<<(H1p + 255) / 256, 256, 0, stream>>>(b1, b1p, H1, H1p);
    k_pad_f32<<<(H2p + 255) / 256, 256, 0, stream>>>(b2, b2p, H2, H2p);
    k_pad_f32<<<(H2p + 255) / 256, 256, 0, stream>>>(bd, bdp, H2, H2p);
    k_zero_bf16<<<(int)(((long)B * H1p + 255) / 256), 256, 0, stream>>>(zero1, (long)B * H1p);
    k_zero_bf16<<<(int)(((long)B * H2p + 255) / 256), 256, 0, stream>>>(h2b0, (long)B * H2p);
  }

  auto mk = [](const bf16_t* A, long lda, const bf16_t* Bt, long ldb, int K,
               const bf16_t* A2, long lda2, const bf16_t* B2t, long ldb2, int K2,
               const float* bias, const float* pre, long ldpre,
               float* outF, long ldoutF, int nvalid,
               bf16_t* outB, long ldoutB, bf16_t* outB2, long ldoutB2,
               int N, int ntx) {
    GP g{A, lda, Bt, ldb, K, A2, lda2, B2t, ldb2, K2, bias, pre, ldpre,
         outF, ldoutF, nvalid, outB, ldoutB, outB2, ldoutB2, N, ntx};
    return g;
  };

  // --- big GEMM: xW1[MT,H1p] = x_bf @ W1 + b1 ---
  {
    GP g = mk(x_bf, F, W1T, F, F,
              nullptr, 0, nullptr, 0, 0,
              b1p, nullptr, 0,
              xW1, H1p, H1p,
              nullptr, 0, nullptr, 0,
              H1p, NB_W1 / 128);
    dim3 grid(NB_W1 / 128, (unsigned)(MT / 128));  // (5, 400)
    k_gemm<128, 128, 2, 4, 256, false><<<grid, 256, 0, stream>>>(g);
  }

  const int NT1 = NB_U1 / 64;  // 9
  const int NT2 = NB_H2 / 64;  // 13

  auto step1GP = [&](int t) {  // h1[t] = tanh(xW1[t] + h1[t-1] @ U1)
    const bf16_t* hprev = t ? (h1seq + (long)(t - 1) * H1p) : zero1;
    long ldh = t ? (long)T * H1p : (long)H1p;
    return mk(hprev, ldh, U1T, H1p, H1p,
              nullptr, 0, nullptr, 0, 0,
              nullptr, xW1 + (long)t * H1p, (long)T * H1p,
              nullptr, 0, 0,
              h1seq + (long)t * H1p, (long)T * H1p, nullptr, 0,
              H1p, NT1);
  };
  auto step2GP = [&](int t) {  // h2 = tanh(h1[t]@W2 + b2 + h2@U2)
    return mk(h1seq + (long)t * H1p, (long)T * H1p, W2T, H1p, H1p,
              h2buf[t & 1], H2p, U2T, H2p, H2p,
              b2p, nullptr, 0,
              nullptr, 0, 0,
              h2buf[(t + 1) & 1], H2p, nullptr, 0,
              H2p, NT2);
  };

  // --- recurrences: step1(0); then pair(layer2(t), layer1(t+1)); then layer2(T-1) ---
  {
    dim3 g1(NT1, B / 32);
    k_gemm<32, 64, 2, 2, 128, true><<<g1, 128, 0, stream>>>(step1GP(0));
    dim3 gp(NT2, B / 32, 2);
    for (int t = 0; t < T - 1; ++t)
      k_gemm_pair<32, 64, 2, 2, 128><<<gp, 128, 0, stream>>>(step2GP(t), step1GP(t + 1));
    dim3 g2(NT2, B / 32);
    k_gemm<32, 64, 2, 2, 128, true><<<g2, 128, 0, stream>>>(step2GP(T - 1));
  }

  // --- dense head: out[256,804] = h2_final @ Wd + bd (final state in h2buf[0]) ---
  {
    GP g = mk(h2buf[0], H2p, WdT, H2p, H2p,
              nullptr, 0, nullptr, 0, 0,
              bdp, nullptr, 0,
              out, H2, H2,
              nullptr, 0, nullptr, 0,
              H2p, NT2);
    dim3 gd(NT2, B / 32);
    k_gemm<32, 64, 2, 2, 128, false><<<gd, 128, 0, stream>>>(g);
  }
}
